// Net_15745350107340
// MI455X (gfx1250) — matrix operand order repaired and checked by reading, not compiled, not run on hardware
//
#include <hip/hip_runtime.h>

#define EMB 32
#define STEPS 3
#define MAXG 1024
#define SLOPE 0.01f
#define ETILES 4   // edge tiles (16 edges each) per wave

typedef __attribute__((ext_vector_type(16))) _Float16 v16h;
typedef __attribute__((ext_vector_type(8)))  _Float16 v8h;
typedef __attribute__((ext_vector_type(8)))  float    v8f;

__device__ __forceinline__ float lrelu(float x) { return x > 0.f ? x : SLOPE * x; }

// monotone float -> uint mapping so atomicMax(u32) == float max; 0 = "empty" sentinel
__device__ __forceinline__ unsigned fenc(float f) {
  unsigned u = __float_as_uint(f);
  return (u & 0x80000000u) ? ~u : (u | 0x80000000u);
}
__device__ __forceinline__ float fdec(unsigned m) {
  return (m & 0x80000000u) ? __uint_as_float(m & 0x7FFFFFFFu) : __uint_as_float(~m);
}
__device__ __forceinline__ float agg_decode(unsigned m) {
  if (m == 0u) return 0.f;                 // empty segment -> 0 (matches isfinite fixup)
  float f = fdec(m);
  return __builtin_isfinite(f) ? f : 0.f;
}

// ---------------------------------------------------------------------------
// Pre-pack B matrices (Wm, Wa chunks, Wfeat) into the per-lane WMMA B layout:
// element e of lane l  <->  K = 8*(l>>4) + (e&7) + 16*(e>>3), col = (l&15) + 16*tile
// Tile order: [0..5]=Wm (step,tile), [6..23]=Wa (step,chunk,tile), [24..29]=Wfeat
// ---------------------------------------------------------------------------
__global__ void pack_b_kernel(const float* __restrict__ Wm,
                              const float* __restrict__ Wa,
                              const float* __restrict__ Wf,
                              _Float16* __restrict__ bpack) {
  int idx = blockIdx.x * blockDim.x + threadIdx.x;
  if (idx >= 30 * 512) return;
  int tile = idx >> 9;
  int within = idx & 511;
  int lane = within >> 4;
  int e = within & 15;
  const float* W;
  int kbase, t;
  if (tile < 6) {
    int s = tile >> 1; t = tile & 1; W = Wm + s * 32 * 32; kbase = 0;
  } else if (tile < 24) {
    int u = tile - 6; int s = u / 6; int rem = u % 6; int chunk = rem >> 1; t = rem & 1;
    W = Wa + s * 96 * 32; kbase = chunk * 32;
  } else {
    int u = tile - 24; int s = u >> 1; t = u & 1; W = Wf + s * 32 * 32; kbase = 0;
  }
  int k = kbase + ((lane >> 4) << 3) + (e & 7) + ((e >> 3) << 4);
  int n = (t << 4) + (lane & 15);
  bpack[idx] = (_Float16)W[k * 32 + n];
}

// x = lrelu(node_feats @ W_embed + b_embed); keep fp32 master + f16 mirror
__global__ void embed_kernel(const float* __restrict__ nf, const float* __restrict__ We,
                             const float* __restrict__ be, float* __restrict__ X,
                             _Float16* __restrict__ XH, int N) {
  long long idx = (long long)blockIdx.x * blockDim.x + threadIdx.x;
  if (idx >= (long long)N * 32) return;
  int n = (int)(idx >> 5);
  int j = (int)(idx & 31);
  float v = lrelu(nf[n] * We[j] + be[j]);
  X[idx] = v;
  XH[idx] = (_Float16)v;
}

__global__ void clear_u32_kernel(unsigned* __restrict__ p, long long n) {
  long long idx = (long long)blockIdx.x * blockDim.x + threadIdx.x;
  if (idx < n) p[idx] = 0u;
}

__global__ void init_graph_kernel(float* __restrict__ pooled, unsigned* __restrict__ gmax,
                                  float* __restrict__ gsum) {
  int idx = blockIdx.x * blockDim.x + threadIdx.x;
  if (idx < MAXG * 32) pooled[idx] = 0.f;
  if (idx < MAXG) { gmax[idx] = 0u; gsum[idx] = 0.f; }
}

// ---------------------------------------------------------------------------
// Fused: msg = lrelu(x[src] @ Wm + bm); agg[dst] = max(agg[dst], msg)
// One wave = ETILES x 16 edges; B stays in registers across tiles; msg never
// hits memory. Full tiles take a branchless (wave-uniform) scatter path.
// ---------------------------------------------------------------------------
__global__ void edge_msg_kernel(const _Float16* __restrict__ xh,
                                const int* __restrict__ ei,        // [2,E]
                                const _Float16* __restrict__ bpack, // 2 tiles
                                const float* __restrict__ bm,      // [32]
                                unsigned* __restrict__ agg, long long E) {
  int lane = threadIdx.x & 31;
  int w = threadIdx.x >> 5;
  long long wbase = ((long long)blockIdx.x * (blockDim.x >> 5) + w) * (16 * ETILES);
  if (wbase >= E) return;                      // wave-uniform: EXEC stays all-ones
  __builtin_prefetch(ei + E + wbase, 0, 0);    // global_prefetch for dst stream
  int row = lane & 15, half = lane >> 4;
  int f0 = lane & 15, f1 = f0 + 16;
  float bb0 = bm[f0], bb1 = bm[f1];
  v16h b0 = *(const v16h*)(bpack + (0 * 32 + lane) * 16);
  v16h b1 = *(const v16h*)(bpack + (1 * 32 + lane) * 16);
  const int* dstp = ei + E;
#pragma unroll
  for (int t = 0; t < ETILES; t++) {
    long long base = wbase + 16 * t;
    if (base >= E) break;                      // wave-uniform
    long long e0 = base + row;
    if (e0 >= E) e0 = E - 1;                   // clamp, keep EXEC full
    int src = ei[e0];
    const _Float16* xr = xh + (long long)src * 32;
    v16h a;
    {
      v8h lo = *(const v8h*)(xr + 8 * half);
      v8h hi = *(const v8h*)(xr + 16 + 8 * half);
#pragma unroll
      for (int i = 0; i < 8; i++) { a[i] = lo[i]; a[8 + i] = hi[i]; }
    }
    v8f c = {};
    v8f d0 = __builtin_amdgcn_wmma_f32_16x16x32_f16(false, a, false, b0, (short)0, c, false, false);
    v8f d1 = __builtin_amdgcn_wmma_f32_16x16x32_f16(false, a, false, b1, (short)0, c, false, false);
    if (base + 16 <= E) {                      // wave-uniform full tile: branchless scatter
#pragma unroll
      for (int r = 0; r < 8; r++) {
        long long em = base + 8 * half + r;
        unsigned* arow = agg + (long long)dstp[em] * 32;
        atomicMax(arow + f0, fenc(lrelu(d0[r] + bb0)));
        atomicMax(arow + f1, fenc(lrelu(d1[r] + bb1)));
      }
    } else {
#pragma unroll
      for (int r = 0; r < 8; r++) {
        long long em = base + 8 * half + r;
        if (em < E) {
          unsigned* arow = agg + (long long)dstp[em] * 32;
          atomicMax(arow + f0, fenc(lrelu(d0[r] + bb0)));
          atomicMax(arow + f1, fenc(lrelu(d1[r] + bb1)));
        }
      }
    }
  }
}

// ---------------------------------------------------------------------------
// x = lrelu([x, xg[batch], agg] @ Wa + ba) + x    (K=96 -> 3 WMMA chunks)
// One wave owns 16 nodes exclusively (safe in-place update of X/XH).
// ---------------------------------------------------------------------------
__global__ void node_update_kernel(float* __restrict__ X, _Float16* __restrict__ XH,
                                   const unsigned* __restrict__ AGG,
                                   const float* __restrict__ XG,
                                   const int* __restrict__ batch,
                                   const _Float16* __restrict__ bpack, // 6 tiles
                                   const float* __restrict__ ba, int N) {
  int lane = threadIdx.x & 31;
  int w = threadIdx.x >> 5;
  long long base = ((long long)blockIdx.x * (blockDim.x >> 5) + w) * 16;
  if (base >= N) return;
  int row = lane & 15, half = lane >> 4;
  long long node = base + row;
  if (node >= N) node = N - 1;
  // chunk0: x (f16 mirror)
  const _Float16* xr = XH + node * 32;
  v16h a0;
  {
    v8h lo = *(const v8h*)(xr + 8 * half);
    v8h hi = *(const v8h*)(xr + 16 + 8 * half);
#pragma unroll
    for (int i = 0; i < 8; i++) { a0[i] = lo[i]; a0[8 + i] = hi[i]; }
  }
  // chunk1: xg[batch[node]]
  const float* gr = XG + (long long)batch[node] * 32;
  v16h a1;
#pragma unroll
  for (int i = 0; i < 8; i++) {
    a1[i]     = (_Float16)gr[8 * half + i];
    a1[8 + i] = (_Float16)gr[16 + 8 * half + i];
  }
  // chunk2: agg (decode monotone-uint max)
  const unsigned* ar = AGG + node * 32;
  v16h a2;
#pragma unroll
  for (int i = 0; i < 8; i++) {
    a2[i]     = (_Float16)agg_decode(ar[8 * half + i]);
    a2[8 + i] = (_Float16)agg_decode(ar[16 + 8 * half + i]);
  }
  v16h b00 = *(const v16h*)(bpack + (0 * 32 + lane) * 16);
  v16h b01 = *(const v16h*)(bpack + (1 * 32 + lane) * 16);
  v16h b10 = *(const v16h*)(bpack + (2 * 32 + lane) * 16);
  v16h b11 = *(const v16h*)(bpack + (3 * 32 + lane) * 16);
  v16h b20 = *(const v16h*)(bpack + (4 * 32 + lane) * 16);
  v16h b21 = *(const v16h*)(bpack + (5 * 32 + lane) * 16);
  v8f d0 = {}, d1 = {};
  d0 = __builtin_amdgcn_wmma_f32_16x16x32_f16(false, a0, false, b00, (short)0, d0, false, false);
  d0 = __builtin_amdgcn_wmma_f32_16x16x32_f16(false, a1, false, b10, (short)0, d0, false, false);
  d0 = __builtin_amdgcn_wmma_f32_16x16x32_f16(false, a2, false, b20, (short)0, d0, false, false);
  d1 = __builtin_amdgcn_wmma_f32_16x16x32_f16(false, a0, false, b01, (short)0, d1, false, false);
  d1 = __builtin_amdgcn_wmma_f32_16x16x32_f16(false, a1, false, b11, (short)0, d1, false, false);
  d1 = __builtin_amdgcn_wmma_f32_16x16x32_f16(false, a2, false, b21, (short)0, d1, false, false);
  int f0 = lane & 15, f1 = f0 + 16;
  float bb0 = ba[f0], bb1 = ba[f1];
  if (base + 16 <= (long long)N) {             // wave-uniform full tile
#pragma unroll
    for (int r = 0; r < 8; r++) {
      long long m = base + 8 * half + r;
      float nv0 = lrelu(d0[r] + bb0) + X[m * 32 + f0];
      X[m * 32 + f0] = nv0;
      XH[m * 32 + f0] = (_Float16)nv0;
      float nv1 = lrelu(d1[r] + bb1) + X[m * 32 + f1];
      X[m * 32 + f1] = nv1;
      XH[m * 32 + f1] = (_Float16)nv1;
    }
  } else {
#pragma unroll
    for (int r = 0; r < 8; r++) {
      long long m = base + 8 * half + r;
      if (m < N) {
        float nv0 = lrelu(d0[r] + bb0) + X[m * 32 + f0];
        X[m * 32 + f0] = nv0;
        XH[m * 32 + f0] = (_Float16)nv0;
        float nv1 = lrelu(d1[r] + bb1) + X[m * 32 + f1];
        X[m * 32 + f1] = nv1;
        XH[m * 32 + f1] = (_Float16)nv1;
      }
    }
  }
}

// segment-softmax phase 1: logit = x.W + b, per-graph max via u32 atomicMax
__global__ void node_logit_max_kernel(const float* __restrict__ X, const int* __restrict__ batch,
                                      const float* __restrict__ W, const float* __restrict__ b,
                                      float* __restrict__ nlog, unsigned* __restrict__ gmax, int N) {
  int n = blockIdx.x * blockDim.x + threadIdx.x;
  if (n >= N) return;
  const float* xr = X + (long long)n * 32;
  float acc = b[0];
#pragma unroll
  for (int k = 0; k < 32; k++) acc += xr[k] * W[k];
  nlog[n] = acc;
  atomicMax(gmax + batch[n], fenc(acc));
}

// phase 2: e = exp(logit - max); nlog := e; gsum += e
__global__ void node_expsum_kernel(float* __restrict__ nlog, const int* __restrict__ batch,
                                   const unsigned* __restrict__ gmax, float* __restrict__ gsum, int N) {
  int n = blockIdx.x * blockDim.x + threadIdx.x;
  if (n >= N) return;
  int g = batch[n];
  float m = fdec(gmax[g]);
  if (!__builtin_isfinite(m)) m = 0.f;
  float e = expf(nlog[n] - m);
  nlog[n] = e;
  atomicAdd(gsum + g, e);
}

// feat = lrelu(x @ Wfeat + bfeat); pooled[g] += (e/s) * feat   (WMMA + f32 atomicAdd)
__global__ void feat_pool_kernel(const _Float16* __restrict__ XH,
                                 const int* __restrict__ batch,
                                 const _Float16* __restrict__ bpack, // 2 tiles
                                 const float* __restrict__ bf,
                                 const float* __restrict__ nlog,     // holds e
                                 const float* __restrict__ gsum,
                                 float* __restrict__ pooled, int N) {
  int lane = threadIdx.x & 31;
  int w = threadIdx.x >> 5;
  long long base = ((long long)blockIdx.x * (blockDim.x >> 5) + w) * 16;
  if (base >= N) return;
  int row = lane & 15, half = lane >> 4;
  long long node = base + row;
  if (node >= N) node = N - 1;
  const _Float16* xr = XH + node * 32;
  v16h a;
  {
    v8h lo = *(const v8h*)(xr + 8 * half);
    v8h hi = *(const v8h*)(xr + 16 + 8 * half);
#pragma unroll
    for (int i = 0; i < 8; i++) { a[i] = lo[i]; a[8 + i] = hi[i]; }
  }
  v16h b0 = *(const v16h*)(bpack + (0 * 32 + lane) * 16);
  v16h b1 = *(const v16h*)(bpack + (1 * 32 + lane) * 16);
  v8f c = {};
  v8f d0 = __builtin_amdgcn_wmma_f32_16x16x32_f16(false, a, false, b0, (short)0, c, false, false);
  v8f d1 = __builtin_amdgcn_wmma_f32_16x16x32_f16(false, a, false, b1, (short)0, c, false, false);
  int f0 = lane & 15, f1 = f0 + 16;
  float bb0 = bf[f0], bb1 = bf[f1];
  if (base + 16 <= (long long)N) {             // wave-uniform full tile
#pragma unroll
    for (int r = 0; r < 8; r++) {
      long long m = base + 8 * half + r;
      int g = batch[m];
      float s = gsum[g];
      float wgt = s > 0.f ? nlog[m] / s : 0.f;
      atomicAdd(pooled + (long long)g * 32 + f0, wgt * lrelu(d0[r] + bb0));
      atomicAdd(pooled + (long long)g * 32 + f1, wgt * lrelu(d1[r] + bb1));
    }
  } else {
#pragma unroll
    for (int r = 0; r < 8; r++) {
      long long m = base + 8 * half + r;
      if (m < N) {
        int g = batch[m];
        float s = gsum[g];
        float wgt = s > 0.f ? nlog[m] / s : 0.f;
        atomicAdd(pooled + (long long)g * 32 + f0, wgt * lrelu(d0[r] + bb0));
        atomicAdd(pooled + (long long)g * 32 + f1, wgt * lrelu(d1[r] + bb1));
      }
    }
  }
}

// xg = lrelu([pooled, xg] @ Wt + bt) + xg   (one 32-thread block per graph)
__global__ void xg_update_kernel(float* __restrict__ XG, const float* __restrict__ pooled,
                                 const float* __restrict__ Wt, const float* __restrict__ bt,
                                 const int* __restrict__ gptr) {
  int g = blockIdx.x;
  if (g >= *gptr) return;
  __shared__ float sxg[32];
  __shared__ float sp[32];
  int j = threadIdx.x;
  sxg[j] = XG[(long long)g * 32 + j];
  sp[j] = pooled[(long long)g * 32 + j];
  __syncthreads();
  float acc = bt[j];
#pragma unroll
  for (int k = 0; k < 32; k++) acc += sp[k] * Wt[k * 32 + j];
#pragma unroll
  for (int k = 0; k < 32; k++) acc += sxg[k] * Wt[(32 + k) * 32 + j];
  XG[(long long)g * 32 + j] = lrelu(acc) + sxg[j];
}

// heads: value + 2-way softmax
__global__ void head_graph_kernel(const float* __restrict__ XG,
                                  const float* __restrict__ Wv, const float* __restrict__ bv,
                                  const float* __restrict__ Wa0, const float* __restrict__ ba0,
                                  float* __restrict__ out, const int* __restrict__ gptr) {
  int g = blockIdx.x * blockDim.x + threadIdx.x;
  int G = *gptr;
  if (g >= G) return;
  const float* xr = XG + (long long)g * 32;
  float v = bv[0], l0 = ba0[0], l1 = ba0[1];
#pragma unroll
  for (int k = 0; k < 32; k++) {
    float x = xr[k];
    v += x * Wv[k];
    l0 += x * Wa0[k * 2 + 0];
    l1 += x * Wa0[k * 2 + 1];
  }
  float m = fmaxf(l0, l1);
  float e0 = expf(l0 - m), e1 = expf(l1 - m);
  float inv = 1.f / (e0 + e1);
  out[g] = v;
  out[G + 2 * g] = e0 * inv;
  out[G + 2 * g + 1] = e1 * inv;
}

__global__ void a1_norm_kernel(const float* __restrict__ nlog, const int* __restrict__ batch,
                               const float* __restrict__ gsum, float* __restrict__ out,
                               const int* __restrict__ gptr, int N) {
  int n = blockIdx.x * blockDim.x + threadIdx.x;
  if (n >= N) return;
  int G = *gptr;
  float s = gsum[batch[n]];
  out[3 * G + n] = s > 0.f ? nlog[n] / s : 0.f;
}

extern "C" void kernel_launch(void* const* d_in, const int* in_sizes, int n_in,
                              void* d_out, int out_size, void* d_ws, size_t ws_size,
                              hipStream_t stream) {
  const float* node_feats = (const float*)d_in[0];
  const int* edge_index = (const int*)d_in[1];
  const int* batch = (const int*)d_in[2];
  const int* gptr = (const int*)d_in[3];
  const float* W_embed = (const float*)d_in[4];
  const float* b_embed = (const float*)d_in[5];
  const float* Wm = (const float*)d_in[6];
  const float* bm = (const float*)d_in[7];
  const float* Wa = (const float*)d_in[8];
  const float* ba = (const float*)d_in[9];
  const float* Wgate = (const float*)d_in[10];
  const float* bgate = (const float*)d_in[11];
  const float* Wfeat = (const float*)d_in[12];
  const float* bfeat = (const float*)d_in[13];
  const float* Wt = (const float*)d_in[14];
  const float* bt = (const float*)d_in[15];
  const float* W_v = (const float*)d_in[16];
  const float* b_v = (const float*)d_in[17];
  const float* W_a0 = (const float*)d_in[18];
  const float* b_a0 = (const float*)d_in[19];
  const float* W_a1 = (const float*)d_in[20];
  const float* b_a1 = (const float*)d_in[21];
  const int N = in_sizes[0];
  const long long E = (long long)in_sizes[1] / 2;
  float* out = (float*)d_out;
  (void)n_in; (void)out_size; (void)ws_size;

  // workspace layout (all from d_ws; 256B aligned slices)
  char* ws = (char*)d_ws;
  size_t off = 0;
  auto take = [&](size_t bytes) { size_t r = off; off += (bytes + 255) & ~(size_t)255; return r; };
  float* X = (float*)(ws + take((size_t)N * 32 * 4));
  _Float16* XH = (_Float16*)(ws + take((size_t)N * 32 * 2));
  unsigned* AGG = (unsigned*)(ws + take((size_t)N * 32 * 4));
  float* NLOG = (float*)(ws + take((size_t)N * 4));
  float* XG = (float*)(ws + take((size_t)MAXG * 32 * 4));
  float* POOLED = (float*)(ws + take((size_t)MAXG * 32 * 4));
  unsigned* GMAX = (unsigned*)(ws + take((size_t)MAXG * 4));
  float* GSUM = (float*)(ws + take((size_t)MAXG * 4));
  _Float16* BPACK = (_Float16*)(ws + take((size_t)30 * 512 * 2));

  const int TB = 256;
  const long long nelem = (long long)N * 32;
  const int elemBlocks = (int)((nelem + TB - 1) / TB);
  const int nodeThreadBlocks = (N + TB - 1) / TB;
  const long long edgeWaves = (E + 16 * ETILES - 1) / (16 * ETILES);
  const int edgeBlocks = (int)((edgeWaves + 7) / 8);
  const long long nodeWaves = ((long long)N + 15) / 16;
  const int nodeBlocks = (int)((nodeWaves + 7) / 8);

  pack_b_kernel<<<(30 * 512 + TB - 1) / TB, TB, 0, stream>>>(Wm, Wa, Wfeat, BPACK);
  embed_kernel<<<elemBlocks, TB, 0, stream>>>(node_feats, W_embed, b_embed, X, XH, N);
  clear_u32_kernel<<<(MAXG * 32 + TB - 1) / TB, TB, 0, stream>>>((unsigned*)XG, MAXG * 32);

  for (int s = 0; s < STEPS; s++) {
    clear_u32_kernel<<<elemBlocks, TB, 0, stream>>>(AGG, nelem);
    edge_msg_kernel<<<edgeBlocks, TB, 0, stream>>>(XH, edge_index, BPACK + (size_t)(s * 2) * 512,
                                                   bm + s * 32, AGG, E);
    node_update_kernel<<<nodeBlocks, TB, 0, stream>>>(X, XH, AGG, XG, batch,
                                                      BPACK + (size_t)(6 + s * 6) * 512,
                                                      ba + s * 32, N);
    init_graph_kernel<<<(MAXG * 32 + TB - 1) / TB, TB, 0, stream>>>(POOLED, GMAX, GSUM);
    node_logit_max_kernel<<<nodeThreadBlocks, TB, 0, stream>>>(X, batch, Wgate + s * 32,
                                                               bgate + s, NLOG, GMAX, N);
    node_expsum_kernel<<<nodeThreadBlocks, TB, 0, stream>>>(NLOG, batch, GMAX, GSUM, N);
    feat_pool_kernel<<<nodeBlocks, TB, 0, stream>>>(XH, batch, BPACK + (size_t)(24 + s * 2) * 512,
                                                    bfeat + s * 32, NLOG, GSUM, POOLED, N);
    xg_update_kernel<<<MAXG, 32, 0, stream>>>(XG, POOLED, Wt + s * 64 * 32, bt + s * 32, gptr);
  }

  head_graph_kernel<<<(MAXG + TB - 1) / TB, TB, 0, stream>>>(XG, W_v, b_v, W_a0, b_a0, out, gptr);
  init_graph_kernel<<<(MAXG * 32 + TB - 1) / TB, TB, 0, stream>>>(POOLED, GMAX, GSUM);
  node_logit_max_kernel<<<nodeThreadBlocks, TB, 0, stream>>>(X, batch, W_a1, b_a1, NLOG, GMAX, N);
  node_expsum_kernel<<<nodeThreadBlocks, TB, 0, stream>>>(NLOG, batch, GMAX, GSUM, N);
  a1_norm_kernel<<<nodeThreadBlocks, TB, 0, stream>>>(NLOG, batch, GSUM, out, gptr, N);
}
